// Model_20925080666713
// MI455X (gfx1250) — matrix operand order repaired and checked by reading, not compiled, not run on hardware
//
#include <hip/hip_runtime.h>
#include <hip/hip_bf16.h>

// ---------------------------------------------------------------------------
// Transformer encoder fwd (B=32,S=512,D=512,H=8,L=4,FFN=1024) on gfx1250.
//  * all GEMMs + attention on v_wmma_f32_16x16x32_f16 (f32 accumulate)
//  * weights converted once/layer to f16, pre-transposed [N,K] -> both GEMM
//    operands are contiguous 128B rows => raw async byte-copies into LDS
//  * GLOBAL_LOAD_ASYNC_TO_LDS_B128 staging (ASYNCcnt) w/ double-buffered LDS
//  * f16 activation twins produced in epilogues (no fp32 reload/convert)
// ---------------------------------------------------------------------------

typedef __attribute__((ext_vector_type(16))) _Float16 v16h;
typedef __attribute__((ext_vector_type(8)))  float    v8f;
typedef __attribute__((ext_vector_type(4)))  float    v4f;
typedef __attribute__((ext_vector_type(4)))  unsigned u32x4;
typedef __attribute__((ext_vector_type(4)))  int      i32x4;
typedef __attribute__((address_space(1))) i32x4 g_i32x4;   // global (AS1)
typedef __attribute__((address_space(3))) i32x4 s_i32x4;   // LDS (AS3)

constexpr int BB = 32, SS = 512, DD = 512, HH = 8, LL = 4, FF = 1024, DH = 64;
constexpr int MROWS = BB * SS;          // 16384
constexpr int MSD   = BB * SS * DD;     // 8388608
constexpr int WPL   = 4 * DD * DD + 2 * DD * FF;  // f16 weights per layer

// -------------------- async global->LDS (gfx1250) --------------------------
#if defined(__has_builtin)
#if __has_builtin(__builtin_amdgcn_global_load_async_to_lds_b128)
#define HAVE_ASYNC_LDS 1
#endif
#if __has_builtin(__builtin_amdgcn_s_wait_asynccnt)
#define HAVE_WAIT_ASYNC 1
#endif
#endif
#ifndef HAVE_ASYNC_LDS
#define HAVE_ASYNC_LDS 0
#endif
#ifndef HAVE_WAIT_ASYNC
#define HAVE_WAIT_ASYNC 0
#endif

__device__ __forceinline__ void async_cp16(const _Float16* g, _Float16* l) {
#if HAVE_ASYNC_LDS
  __builtin_amdgcn_global_load_async_to_lds_b128((g_i32x4*)g, (s_i32x4*)l, 0, 0);
#else
  *(u32x4*)l = *(const u32x4*)g;
#endif
}

__device__ __forceinline__ void async_wait0() {
#if HAVE_ASYNC_LDS
#if HAVE_WAIT_ASYNC
  __builtin_amdgcn_s_wait_asynccnt(0);
#else
  asm volatile("s_wait_asynccnt 0x0" ::: "memory");
#endif
#endif
}

// ---------------------------------------------------------------------------
// WMMA fragment loaders from LDS (CDNA5 ISA 7.12.2 layouts).
// A (16x32 f16): lane->M=lane&15, laneHalf->K+8; VGPR v: K=(v&4)*4+half*8+(v&3)*2
// B (32x16 f16; LDS stored [n][k]): lane->N=lane&15, laneHalf->K+8; v: K=(v&4)*4+(v&3)*2 (same as A)
// C/D (16x16 f32): VGPR r -> M = r + laneHalf*8, N = lane&15
// ---------------------------------------------------------------------------
__device__ __forceinline__ v16h load_a_frag(const _Float16* base, int ld,
                                            int lane) {
  union { v16h v; unsigned u[8]; } f;
  const _Float16* p = base + (lane & 15) * ld;
  const int klane = ((lane >> 4) & 1) * 8;
  #pragma unroll
  for (int vv = 0; vv < 8; ++vv) {
    int k = ((vv & 4) << 2) + klane + ((vv & 3) << 1);
    f.u[vv] = *(const unsigned*)(p + k);
  }
  return f.v;
}

__device__ __forceinline__ v16h load_b_frag(const _Float16* base, int ld,
                                            int lane) {
  union { v16h v; unsigned u[8]; } f;
  const _Float16* p = base + (lane & 15) * ld;
  const int kg = ((lane >> 4) & 1) * 8;
  #pragma unroll
  for (int vv = 0; vv < 8; ++vv) f.u[vv] = *(const unsigned*)(p + ((vv & 4) << 2) + kg + ((vv & 3) << 1));
  return f.v;
}

__device__ __forceinline__ v8f wmma_f16(v16h a, v16h b, v8f c) {
  return __builtin_amdgcn_wmma_f32_16x16x32_f16(false, a, false, b, (short)0, c,
                                                false, false);
}

// ---------------------------------------------------------------------------
// y32 = x + pe ; y16 = (f16)(x + pe)
// ---------------------------------------------------------------------------
__global__ __launch_bounds__(256) void add_pe_k(const float* __restrict__ x,
                                                const float* __restrict__ pe,
                                                float* __restrict__ y32,
                                                _Float16* __restrict__ y16) {
  int i = (blockIdx.x * 256 + threadIdx.x) * 4;
  v4f a = *(const v4f*)(x + i);
  v4f p = *(const v4f*)(pe + (i & (SS * DD - 1)));
  v4f r = a + p;
  *(v4f*)(y32 + i) = r;
  union { unsigned long long u; _Float16 h[4]; } ph;
  ph.h[0] = (_Float16)r[0]; ph.h[1] = (_Float16)r[1];
  ph.h[2] = (_Float16)r[2]; ph.h[3] = (_Float16)r[3];
  *(unsigned long long*)(y16 + i) = ph.u;
}

// ---------------------------------------------------------------------------
// Weight convert+transpose: fp32 W[K,N] -> f16 Wt[N,K] (64x64 LDS tiles)
// ---------------------------------------------------------------------------
__global__ __launch_bounds__(256) void convT_k(const float* __restrict__ W,
                                               _Float16* __restrict__ Wt,
                                               int K, int N) {
  __shared__ _Float16 sh[64][65];
  const int k0 = blockIdx.y * 64, n0 = blockIdx.x * 64;
  const int tx = threadIdx.x & 63, ty = threadIdx.x >> 6;
  #pragma unroll
  for (int r = ty; r < 64; r += 4)
    sh[r][tx] = (_Float16)W[(size_t)(k0 + r) * N + n0 + tx];
  __syncthreads();
  #pragma unroll
  for (int r = ty; r < 64; r += 4)
    Wt[(size_t)(n0 + r) * K + k0 + tx] = sh[tx][r];
}

// ---------------------------------------------------------------------------
// C = act( A[M,K](f16) @ Wt[N,K](f16)^T + bias (+ res32) )
// 256 thr = 8 waves; block tile 128x128, BK=64; wave tile 32x64 (2x4 WMMA).
// Double-buffered LDS, async global->LDS staging.
// ---------------------------------------------------------------------------
template <bool RELU, bool RES, bool OUT32, bool OUT16>
__global__ __launch_bounds__(256) void gemm_f16(
    const _Float16* __restrict__ A, const _Float16* __restrict__ Wt,
    const float* __restrict__ bias, const float* __restrict__ res,
    float* __restrict__ C32, _Float16* __restrict__ C16, float oscale,
    int M, int N, int K) {
  constexpr int LDA = 72;  // halves; 144B row pitch (16B aligned, 36-bank step)
  __shared__ __align__(16) _Float16 As[2][128 * LDA];
  __shared__ __align__(16) _Float16 Bs[2][128 * LDA];

  const int tid  = threadIdx.x;
  const int lane = tid & 31;
  const int wid  = tid >> 5;
  const int wm   = (wid >> 1) * 32;
  const int wn   = (wid & 1) * 64;
  const int m0   = blockIdx.y * 128;
  const int n0   = blockIdx.x * 128;

  const _Float16* ag = A  + (size_t)m0 * K;  // tile row stride K halves
  const _Float16* bg = Wt + (size_t)n0 * K;

  v8f acc[2][4];
  #pragma unroll
  for (int mi = 0; mi < 2; ++mi)
    #pragma unroll
    for (int ni = 0; ni < 4; ++ni) acc[mi][ni] = (v8f){};

  // stage tile pair (128 rows x 64 halves each) into buffer `buf`
  auto stage = [&](int buf, int k0) {
    #pragma unroll
    for (int i = 0; i < 4; ++i) {          // 1024 16B-segments per matrix
      const int s = tid + i * 256;
      const int row = s >> 3, seg = (s & 7) * 8;
      async_cp16(ag + (size_t)row * K + k0 + seg, &As[buf][row * LDA + seg]);
      async_cp16(bg + (size_t)row * K + k0 + seg, &Bs[buf][row * LDA + seg]);
    }
    if (k0 + 128 < K) {  // warm L2 two stages ahead
      __builtin_prefetch(ag + (size_t)(tid >> 1) * K + k0 + 128 + (tid & 1) * 32, 0, 0);
      __builtin_prefetch(bg + (size_t)(tid >> 1) * K + k0 + 128 + (tid & 1) * 32, 0, 0);
    }
  };

  stage(0, 0);
  async_wait0();
  __syncthreads();

  const int nk = K >> 6;
  for (int t = 0; t < nk; ++t) {
    const int buf = t & 1;
    if (t + 1 < nk) stage(buf ^ 1, (t + 1) * 64);
    #pragma unroll
    for (int kk = 0; kk < 64; kk += 32) {
      v16h af[2], bf[4];
      #pragma unroll
      for (int mi = 0; mi < 2; ++mi)
        af[mi] = load_a_frag(&As[buf][(wm + mi * 16) * LDA + kk], LDA, lane);
      #pragma unroll
      for (int ni = 0; ni < 4; ++ni)
        bf[ni] = load_b_frag(&Bs[buf][(wn + ni * 16) * LDA + kk], LDA, lane);
      #pragma unroll
      for (int mi = 0; mi < 2; ++mi)
        #pragma unroll
        for (int ni = 0; ni < 4; ++ni)
          acc[mi][ni] = wmma_f16(af[mi], bf[ni], acc[mi][ni]);
    }
    async_wait0();
    __syncthreads();
  }

  // epilogue
  const int half = (lane >> 4) & 1;
  const int ncol = lane & 15;
  #pragma unroll
  for (int mi = 0; mi < 2; ++mi) {
    #pragma unroll
    for (int ni = 0; ni < 4; ++ni) {
      const int n  = n0 + wn + ni * 16 + ncol;
      const float bv = bias[n];
      const int mb = m0 + wm + mi * 16 + half * 8;
      #pragma unroll
      for (int r = 0; r < 8; ++r) {
        const size_t idx = (size_t)(mb + r) * N + n;
        float val = acc[mi][ni][r] + bv;
        if (RES)  val += res[idx];
        if (RELU) val = fmaxf(val, 0.0f);
        if (OUT32) C32[idx] = val;
        if (OUT16) C16[idx] = (_Float16)(val * oscale);
      }
    }
  }
}

// ---------------------------------------------------------------------------
// Flash attention per head (raw-view: head = contiguous S x 64 block).
// f16 in (Q pre-scaled by 1/8 at projection), f16 out. Grid (B*H, S/64), 128t.
// ---------------------------------------------------------------------------
__global__ __launch_bounds__(128) void attn_head(const _Float16* __restrict__ Q,
                                                 const _Float16* __restrict__ Kg,
                                                 const _Float16* __restrict__ V,
                                                 _Float16* __restrict__ O) {
  constexpr int LDQ = 72;  // 144B pitch
  __shared__ __align__(16) _Float16 Qs[64 * LDQ];
  __shared__ __align__(16) _Float16 Ks[64 * LDQ];
  __shared__ __align__(16) _Float16 Vt[64 * LDQ];   // [dh][key]
  __shared__ __align__(16) _Float16 Ps[4][16 * LDQ];

  const int head = blockIdx.x;
  const int q0   = blockIdx.y * 64;
  const int tid  = threadIdx.x;
  const int lane = tid & 31;
  const int wid  = tid >> 5;

  const size_t hb = (size_t)head * SS * DH;
  const _Float16* Qh = Q + hb;
  const _Float16* Kh = Kg + hb;
  const _Float16* Vh = V + hb;
  _Float16* Oh = O + hb;

  // async-stage Q tile (64 rows x 64 halves); waited inside first kv iter
  #pragma unroll
  for (int i = 0; i < 4; ++i) {
    const int s = tid + i * 128;             // 512 16B segments
    const int row = s >> 3, seg = (s & 7) * 8;
    async_cp16(Qh + (size_t)(q0 + row) * DH + seg, &Qs[row * LDQ + seg]);
  }

  v8f o[4];
  #pragma unroll
  for (int ni = 0; ni < 4; ++ni) o[ni] = (v8f){};
  float mrow[8], lrow[8];
  #pragma unroll
  for (int r = 0; r < 8; ++r) { mrow[r] = -1e30f; lrow[r] = 0.0f; }

  for (int kv0 = 0; kv0 < SS; kv0 += 64) {
    __syncthreads();  // prior compute finished reading Ks/Vt
    #pragma unroll
    for (int i = 0; i < 4; ++i) {            // async K tile, row-major
      const int s = tid + i * 128;
      const int row = s >> 3, seg = (s & 7) * 8;
      async_cp16(Kh + (size_t)(kv0 + row) * DH + seg, &Ks[row * LDQ + seg]);
    }
    {  // V tile transposed via VGPRs (async cannot transpose)
      const int row = tid >> 1, cb = (tid & 1) * 32;
      union { u32x4 q[4]; _Float16 h[32]; } tv;
      const _Float16* vp = Vh + (size_t)(kv0 + row) * DH + cb;
      #pragma unroll
      for (int j = 0; j < 4; ++j) tv.q[j] = *(const u32x4*)(vp + j * 8);
      #pragma unroll
      for (int j = 0; j < 32; ++j) Vt[(cb + j) * LDQ + row] = tv.h[j];
    }
    async_wait0();
    __syncthreads();

    // scores 16x64 per wave
    v8f s[4];
    #pragma unroll
    for (int ni = 0; ni < 4; ++ni) s[ni] = (v8f){};
    #pragma unroll
    for (int kk = 0; kk < 64; kk += 32) {
      v16h a = load_a_frag(&Qs[(wid * 16) * LDQ + kk], LDQ, lane);
      #pragma unroll
      for (int ni = 0; ni < 4; ++ni)
        s[ni] = wmma_f16(a, load_b_frag(&Ks[(ni * 16) * LDQ + kk], LDQ, lane),
                         s[ni]);
    }

    // online softmax (C-layout rows; 16-lane halves hold independent rows)
    #pragma unroll
    for (int r = 0; r < 8; ++r) {
      float mx = fmaxf(fmaxf(s[0][r], s[1][r]), fmaxf(s[2][r], s[3][r]));
      mx = fmaxf(mx, __shfl_xor(mx, 1, 32));
      mx = fmaxf(mx, __shfl_xor(mx, 2, 32));
      mx = fmaxf(mx, __shfl_xor(mx, 4, 32));
      mx = fmaxf(mx, __shfl_xor(mx, 8, 32));
      const float mnew  = fmaxf(mrow[r], mx);
      const float alpha = __expf(mrow[r] - mnew);
      float psum = 0.0f;
      #pragma unroll
      for (int ni = 0; ni < 4; ++ni) {
        const float p = __expf(s[ni][r] - mnew);
        s[ni][r] = p;
        psum += p;
        o[ni][r] *= alpha;
      }
      psum += __shfl_xor(psum, 1, 32);
      psum += __shfl_xor(psum, 2, 32);
      psum += __shfl_xor(psum, 4, 32);
      psum += __shfl_xor(psum, 8, 32);
      lrow[r] = lrow[r] * alpha + psum;
      mrow[r] = mnew;
    }

    // P -> LDS in A-matrix layout, then O += P @ V
    {
      const int prow = ((lane >> 4) & 1) * 8;
      const int pcol = lane & 15;
      #pragma unroll
      for (int ni = 0; ni < 4; ++ni)
        #pragma unroll
        for (int r = 0; r < 8; ++r)
          Ps[wid][(r + prow) * LDQ + ni * 16 + pcol] = (_Float16)s[ni][r];
    }
    #pragma unroll
    for (int kk = 0; kk < 64; kk += 32) {
      v16h a = load_a_frag(&Ps[wid][kk], LDQ, lane);
      #pragma unroll
      for (int ni = 0; ni < 4; ++ni)
        o[ni] = wmma_f16(a, load_b_frag(&Vt[(ni * 16) * LDQ + kk], LDQ, lane),
                         o[ni]);
    }
  }

  const int rowb = q0 + wid * 16 + ((lane >> 4) & 1) * 8;
  const int ncol = lane & 15;
  float inv[8];
  #pragma unroll
  for (int r = 0; r < 8; ++r) inv[r] = 1.0f / lrow[r];
  #pragma unroll
  for (int ni = 0; ni < 4; ++ni)
    #pragma unroll
    for (int r = 0; r < 8; ++r)
      Oh[(size_t)(rowb + r) * DH + ni * 16 + ncol] =
          (_Float16)(o[ni][r] * inv[r]);
}

// ---------------------------------------------------------------------------
// LayerNorm over last dim (512); one wave/row; optional f16 twin output.
// ---------------------------------------------------------------------------
template <bool W16>
__global__ __launch_bounds__(256) void layernorm_k(const float* __restrict__ X,
                                                   const float* __restrict__ g,
                                                   const float* __restrict__ b,
                                                   float* __restrict__ Y32,
                                                   _Float16* __restrict__ Y16) {
  const int row  = blockIdx.x * 8 + (threadIdx.x >> 5);
  const int lane = threadIdx.x & 31;
  const float* x = X + (size_t)row * DD;
  float v[16], s = 0.0f, ss = 0.0f;
  #pragma unroll
  for (int i = 0; i < 16; ++i) {
    v[i] = x[lane + i * 32];
    s += v[i];
    ss += v[i] * v[i];
  }
  #pragma unroll
  for (int off = 16; off >= 1; off >>= 1) {
    s  += __shfl_xor(s, off, 32);
    ss += __shfl_xor(ss, off, 32);
  }
  const float mean = s * (1.0f / DD);
  const float var  = ss * (1.0f / DD) - mean * mean;
  const float rstd = rsqrtf(var + 1e-5f);
  float* y = Y32 + (size_t)row * DD;
  _Float16* y16 = W16 ? Y16 + (size_t)row * DD : nullptr;
  #pragma unroll
  for (int i = 0; i < 16; ++i) {
    const int c = lane + i * 32;
    const float r = (v[i] - mean) * rstd * g[c] + b[c];
    y[c] = r;
    if (W16) y16[c] = (_Float16)r;
  }
}

// ---------------------------------------------------------------------------
// Host orchestration
// ---------------------------------------------------------------------------
extern "C" void kernel_launch(void* const* d_in, const int* in_sizes, int n_in,
                              void* d_out, int out_size, void* d_ws,
                              size_t ws_size, hipStream_t stream) {
  const float* x    = (const float*)d_in[0];
  const float* pe   = (const float*)d_in[1];
  const float* Wq   = (const float*)d_in[2];
  const float* bq   = (const float*)d_in[3];
  const float* Wk   = (const float*)d_in[4];
  const float* bk   = (const float*)d_in[5];
  const float* Wv   = (const float*)d_in[6];
  const float* bv   = (const float*)d_in[7];
  const float* Wo   = (const float*)d_in[8];
  const float* bo   = (const float*)d_in[9];
  const float* ln1g = (const float*)d_in[10];
  const float* ln1b = (const float*)d_in[11];
  const float* W1   = (const float*)d_in[12];
  const float* b1   = (const float*)d_in[13];
  const float* W2   = (const float*)d_in[14];
  const float* b2   = (const float*)d_in[15];
  const float* ln2g = (const float*)d_in[16];
  const float* ln2b = (const float*)d_in[17];
  float* out = (float*)d_out;

  // workspace: 3 fp32 activation buffers + f16 pool
  float* ws = (float*)d_ws;
  float* xb = ws;                 // residual stream (fp32)
  float* ab = ws + 1ll * MSD;     // post-LN1 (fp32, FFN2 residual)
  float* tb = ws + 2ll * MSD;     // pre-LN temp (fp32)
  _Float16* f16p = (_Float16*)(ws + 3ll * MSD);
  _Float16* x16 = f16p;                 // activations f16 twin
  _Float16* q16 = f16p + 1ll * MSD;
  _Float16* k16 = f16p + 2ll * MSD;
  _Float16* v16 = f16p + 3ll * MSD;
  _Float16* c16 = f16p + 4ll * MSD;     // attention context
  _Float16* w16 = f16p + 5ll * MSD;     // all layers' weights (WPL*LL = MSD)
  _Float16* a16 = q16;                  // Q dead after attention
  _Float16* h16 = k16;                  // FFN hidden spans k16+v16 (2*MSD)

  const dim3 blk256(256), blk128(128);
  const dim3 gD(DD / 128, MROWS / 128);
  const dim3 gF(FF / 128, MROWS / 128);
  const dim3 gAttn(BB * HH, SS / 64);
  const dim3 gLN(MROWS / 8);

  // one-time per call: weight convert + transpose to f16 [N,K]
  for (int l = 0; l < LL; ++l) {
    _Float16* wl = w16 + (size_t)l * WPL;
    convT_k<<<dim3(DD / 64, DD / 64), blk256, 0, stream>>>(
        Wq + (size_t)l * DD * DD, wl + 0ll * DD * DD, DD, DD);
    convT_k<<<dim3(DD / 64, DD / 64), blk256, 0, stream>>>(
        Wk + (size_t)l * DD * DD, wl + 1ll * DD * DD, DD, DD);
    convT_k<<<dim3(DD / 64, DD / 64), blk256, 0, stream>>>(
        Wv + (size_t)l * DD * DD, wl + 2ll * DD * DD, DD, DD);
    convT_k<<<dim3(DD / 64, DD / 64), blk256, 0, stream>>>(
        Wo + (size_t)l * DD * DD, wl + 3ll * DD * DD, DD, DD);
    convT_k<<<dim3(FF / 64, DD / 64), blk256, 0, stream>>>(
        W1 + (size_t)l * DD * FF, wl + 4ll * DD * DD, DD, FF);
    convT_k<<<dim3(DD / 64, FF / 64), blk256, 0, stream>>>(
        W2 + (size_t)l * FF * DD, wl + 4ll * DD * DD + (size_t)FF * DD, FF, DD);
  }

  add_pe_k<<<MSD / (256 * 4), blk256, 0, stream>>>(x, pe, xb, x16);

  for (int l = 0; l < LL; ++l) {
    _Float16* wl  = w16 + (size_t)l * WPL;
    _Float16* wqt = wl;
    _Float16* wkt = wl + 1ll * DD * DD;
    _Float16* wvt = wl + 2ll * DD * DD;
    _Float16* wot = wl + 3ll * DD * DD;
    _Float16* w1t = wl + 4ll * DD * DD;
    _Float16* w2t = wl + 4ll * DD * DD + (size_t)FF * DD;

    // projections -> f16 (Q pre-scaled by dh^-0.5)
    gemm_f16<false, false, false, true><<<gD, blk256, 0, stream>>>(
        x16, wqt, bq + l * DD, nullptr, nullptr, q16, 0.125f, MROWS, DD, DD);
    gemm_f16<false, false, false, true><<<gD, blk256, 0, stream>>>(
        x16, wkt, bk + l * DD, nullptr, nullptr, k16, 1.0f, MROWS, DD, DD);
    gemm_f16<false, false, false, true><<<gD, blk256, 0, stream>>>(
        x16, wvt, bv + l * DD, nullptr, nullptr, v16, 1.0f, MROWS, DD, DD);

    attn_head<<<gAttn, blk128, 0, stream>>>(q16, k16, v16, c16);

    // t = ctx @ Wo + bo + x ; a = LN1(t) (fp32 + f16)
    gemm_f16<false, true, true, false><<<gD, blk256, 0, stream>>>(
        c16, wot, bo + l * DD, xb, tb, nullptr, 1.0f, MROWS, DD, DD);
    layernorm_k<true><<<gLN, blk256, 0, stream>>>(tb, ln1g + l * DD,
                                                  ln1b + l * DD, ab, a16);
    // h = relu(a @ W1 + b1) -> f16
    gemm_f16<true, false, false, true><<<gF, blk256, 0, stream>>>(
        a16, w1t, b1 + l * FF, nullptr, nullptr, h16, 1.0f, MROWS, FF, DD);
    // t = h @ W2 + b2 + a ; out = LN2(t)
    gemm_f16<false, true, true, false><<<gD, blk256, 0, stream>>>(
        h16, w2t, b2 + l * DD, ab, tb, nullptr, 1.0f, MROWS, DD, FF);
    if (l == LL - 1) {
      layernorm_k<false><<<gLN, blk256, 0, stream>>>(tb, ln2g + l * DD,
                                                     ln2b + l * DD, out,
                                                     nullptr);
    } else {
      layernorm_k<true><<<gLN, blk256, 0, stream>>>(tb, ln2g + l * DD,
                                                    ln2b + l * DD, xb, x16);
    }
  }
}
